// Attention_61383672594598
// MI455X (gfx1250) — compile-verified, matrix operand order checked by reading, not run on hardware
//
#include <hip/hip_runtime.h>

#define N_NODES 12288
#define IN_F    256
#define OUT_F   128
#define ALPHA   0.2f

typedef __attribute__((ext_vector_type(16))) __bf16 v16bf;
typedef __attribute__((ext_vector_type(8)))  __bf16 v8bf;
typedef __attribute__((ext_vector_type(8)))  float  v8f;

#define NEG_INF (-__builtin_inff())

// ---------------------------------------------------------------------------
// Kernel 1: h = x @ W  (fp32 in, bf16 WMMA, fp32 accumulate), store h
// TRANSPOSED in bf16: hbT[c * N + n] = h[n][c].  One wave per 16x16 tile.
// ---------------------------------------------------------------------------
__global__ __launch_bounds__(32) void h_gemm_kernel(const float* __restrict__ x,
                                                    const float* __restrict__ W,
                                                    __bf16* __restrict__ hbT) {
  const int lane = threadIdx.x;
  const int mr = lane & 15;   // A: row / B: col / C: col within tile
  const int hk = lane >> 4;   // K half selector
  const int r0 = blockIdx.x * 16;   // node-row tile
  const int c0 = blockIdx.y * 16;   // feature-col tile

  v8f acc = {0.f, 0.f, 0.f, 0.f, 0.f, 0.f, 0.f, 0.f};

  for (int k0 = 0; k0 < IN_F; k0 += 32) {
    // A operand: x[r0+mr, k0 + {hk*8..hk*8+7, 16+hk*8..16+hk*8+7}]
    const float* xr = x + (size_t)(r0 + mr) * IN_F + (k0 + hk * 8);
    float af[16];
    *(float4*)&af[0]  = *(const float4*)(xr);
    *(float4*)&af[4]  = *(const float4*)(xr + 4);
    *(float4*)&af[8]  = *(const float4*)(xr + 16);
    *(float4*)&af[12] = *(const float4*)(xr + 20);
    v16bf av;
#pragma unroll
    for (int t = 0; t < 16; ++t) av[t] = (__bf16)af[t];

    // B operand: W[k0 + krun, c0 + mr]  (K-strided scalar loads; W is tiny, L0-hot)
    const float* wc = W + (size_t)(k0 + hk * 8) * OUT_F + (c0 + mr);
    v16bf bv;
#pragma unroll
    for (int t = 0; t < 8; ++t) {
      bv[t]     = (__bf16)wc[(size_t)t * OUT_F];
      bv[t + 8] = (__bf16)wc[(size_t)(t + 16) * OUT_F];
    }

    acc = __builtin_amdgcn_wmma_f32_16x16x32_bf16(false, av, false, bv,
                                                  (short)0, acc, false, false);
  }

  // C layout: acc[r] = h[r0 + hk*8 + r, c0 + mr]  -> store transposed, 16B store
  v8bf o;
#pragma unroll
  for (int r = 0; r < 8; ++r) o[r] = (__bf16)acc[r];
  *(v8bf*)(hbT + (size_t)(c0 + mr) * (size_t)N_NODES + (r0 + hk * 8)) = o;
}

// ---------------------------------------------------------------------------
// Kernel 2: s1 = h @ a1, s2 = h @ a2 from transposed bf16 h (coalesced in n).
// ---------------------------------------------------------------------------
__global__ __launch_bounds__(256) void score_kernel(const __bf16* __restrict__ hbT,
                                                    const float* __restrict__ a1,
                                                    const float* __restrict__ a2,
                                                    float* __restrict__ s1,
                                                    float* __restrict__ s2) {
  const int n = blockIdx.x * blockDim.x + threadIdx.x;
  if (n >= N_NODES) return;
  float t1 = 0.f, t2 = 0.f;
#pragma unroll 4
  for (int c = 0; c < OUT_F; ++c) {
    float hv = (float)hbT[(size_t)c * N_NODES + n];
    t1 += hv * a1[c];
    t2 += hv * a2[c];
  }
  s1[n] = t1;
  s2[n] = t2;
}

// ---------------------------------------------------------------------------
// Kernel 3: fused masked-softmax(e) @ h, flash style.  4 waves / block, each
// wave owns 16 rows x 128 out cols; streams j in chunks of 32 (WMMA K).
//
// Key trick: the running softmax max is taken over UNMASKED leaky-relu scores
// (exact: a uniformly larger max cancels in normalization), so no -inf
// plumbing; the adjacency mask is applied as a select on p only.  The
// accumulator rescale (64 muls + 8 shuffles) runs under a wave-uniform branch
// that fires only when some row's max actually increased (~9 of 384 chunks).
// ---------------------------------------------------------------------------
__global__ __launch_bounds__(128) void gat_kernel(const int* __restrict__ adj,
                                                  const __bf16* __restrict__ hbT,
                                                  const float* __restrict__ s1,
                                                  const float* __restrict__ s2,
                                                  float* __restrict__ out) {
  const int wave = threadIdx.x >> 5;
  const int lane = threadIdx.x & 31;
  const int mr = lane & 15;
  const int hk = lane >> 4;
  const int r0 = (blockIdx.x * 4 + wave) * 16;
  const int row = r0 + mr;                 // row whose softmax state this lane tracks
  const float s2row = s2[row];
  const int* __restrict__ arow = adj + (size_t)row * N_NODES;

  float m_run = NEG_INF;
  float l_run = 0.f;
  v8f acc[8];
#pragma unroll
  for (int s = 0; s < 8; ++s) acc[s] = (v8f){0.f, 0.f, 0.f, 0.f, 0.f, 0.f, 0.f, 0.f};

  for (int j0 = 0; j0 < N_NODES; j0 += 32) {
    const int kA = j0 + hk * 8;        // first contiguous K-run of this lane
    const int kB = j0 + 16 + hk * 8;   // second contiguous K-run

    float s1v[16];
    *(float4*)&s1v[0]  = *(const float4*)(s1 + kA);
    *(float4*)&s1v[4]  = *(const float4*)(s1 + kA + 4);
    *(float4*)&s1v[8]  = *(const float4*)(s1 + kB);
    *(float4*)&s1v[12] = *(const float4*)(s1 + kB + 4);

    int adjv[16];
    *(int4*)&adjv[0]  = *(const int4*)(arow + kA);
    *(int4*)&adjv[4]  = *(const int4*)(arow + kA + 4);
    *(int4*)&adjv[8]  = *(const int4*)(arow + kB);
    *(int4*)&adjv[12] = *(const int4*)(arow + kB + 4);

    // e = leakyrelu(s1[j] + s2[i])  (UNMASKED; leaky(v) = max(v, 0.2v))
    float e[16];
#pragma unroll
    for (int t = 0; t < 16; ++t) {
      float v = s1v[t] + s2row;
      e[t] = fmaxf(v, ALPHA * v);
    }

    // chunk max over this row's 32 columns (lane pair L, L^16)
    float cmax = e[0];
#pragma unroll
    for (int t = 1; t < 16; ++t) cmax = fmaxf(cmax, e[t]);
    cmax = fmaxf(cmax, __shfl_xor(cmax, 16, 32));

    // rare path: some row's running max increased -> rescale state + acc
    if (__any(cmax > m_run)) {
      const float m_new = fmaxf(m_run, cmax);
      const float scale = __expf(m_run - m_new);   // first chunk: exp(-inf)=0
      l_run *= scale;
      m_run = m_new;
      const int base = hk * 8;
      float sc[8];
#pragma unroll
      for (int r = 0; r < 8; ++r) sc[r] = __shfl(scale, base + r, 32);
#pragma unroll
      for (int s = 0; s < 8; ++s)
#pragma unroll
        for (int r = 0; r < 8; ++r) acc[s][r] *= sc[r];
    }

    // p = exp(e - m) masked by adjacency; accumulate row sum
    float p[16];
    float psum = 0.f;
#pragma unroll
    for (int t = 0; t < 16; ++t) {
      float pv = __expf(e[t] - m_run);
      pv = (adjv[t] > 0) ? pv : 0.f;
      p[t] = pv;
      psum += pv;
    }
    psum += __shfl_xor(psum, 16, 32);
    l_run += psum;

    // A operand: probabilities in bf16, already in A-layout order
    v16bf pa;
#pragma unroll
    for (int t = 0; t < 16; ++t) pa[t] = (__bf16)p[t];

    // 8 output slabs of 16 features: B = hbT[c, j0..j0+31], two b128s per slab
#pragma unroll
    for (int s = 0; s < 8; ++s) {
      const __bf16* hcol = hbT + (size_t)(s * 16 + mr) * (size_t)N_NODES;
      v8bf b0 = *(const v8bf*)(hcol + kA);
      v8bf b1 = *(const v8bf*)(hcol + kB);
      v16bf bv;
#pragma unroll
      for (int t = 0; t < 8; ++t) { bv[t] = b0[t]; bv[t + 8] = b1[t]; }
      acc[s] = __builtin_amdgcn_wmma_f32_16x16x32_bf16(false, pa, false, bv,
                                                       (short)0, acc[s], false, false);
    }
  }

  // epilogue: divide by softmax denominator of each C row and store
  const int base = hk * 8;
  float lr[8];
#pragma unroll
  for (int r = 0; r < 8; ++r) {
    float lv = __shfl(l_run, base + r, 32);
    lr[r] = (lv > 0.f) ? (1.f / lv) : 0.f;
  }
#pragma unroll
  for (int s = 0; s < 8; ++s)
#pragma unroll
    for (int r = 0; r < 8; ++r)
      out[(size_t)(r0 + base + r) * OUT_F + s * 16 + mr] = acc[s][r] * lr[r];
}

// ---------------------------------------------------------------------------
extern "C" void kernel_launch(void* const* d_in, const int* in_sizes, int n_in,
                              void* d_out, int out_size, void* d_ws, size_t ws_size,
                              hipStream_t stream) {
  const float* x   = (const float*)d_in[0];   // [N, IN_F]
  const int*   adj = (const int*)d_in[1];     // [N, N]
  const float* W   = (const float*)d_in[2];   // [IN_F, OUT_F]
  const float* a1  = (const float*)d_in[3];   // [OUT_F]
  const float* a2  = (const float*)d_in[4];   // [OUT_F]
  float* out = (float*)d_out;                 // [N, OUT_F]

  char* ws = (char*)d_ws;
  __bf16* hbT = (__bf16*)ws;                                   // [OUT_F, N] bf16
  size_t hbT_bytes = (size_t)OUT_F * N_NODES * sizeof(__bf16); // ~3.1 MB
  float* s1 = (float*)(ws + hbT_bytes);
  float* s2 = s1 + N_NODES;

  h_gemm_kernel<<<dim3(N_NODES / 16, OUT_F / 16), 32, 0, stream>>>(x, W, hbT);
  score_kernel<<<N_NODES / 256, 256, 0, stream>>>(hbT, a1, a2, s1, s2);
  gat_kernel<<<N_NODES / 64, 128, 0, stream>>>(adj, hbT, s1, s2, out);
}
